// SubcenterArcface_88751204204844
// MI455X (gfx1250) — matrix operand order repaired and checked by reading, not compiled, not run on hardware
//
#include <hip/hip_runtime.h>

typedef __attribute__((ext_vector_type(16))) _Float16 v16h;
typedef __attribute__((ext_vector_type(8)))  _Float16 v8h;
typedef __attribute__((ext_vector_type(4)))  _Float16 v4h;
typedef __attribute__((ext_vector_type(8)))  float    v8f;

#define DIM 512
#define OUT_CLASSES 100000
#define KSUB 3
#define NCOL (OUT_CLASSES * KSUB)   // 300000
#define TILE_M 128
#define TILE_N 96                   // 32 whole classes per tile
#define LDS_STRIDE 520              // halves, padded (1040B rows: 16B aligned, bank-staggered)
#define A_HALVES (TILE_M * LDS_STRIDE)
#define B_HALVES (TILE_N * LDS_STRIDE)
#define C_STRIDE 100                // f32 stride for epilogue staging

__device__ __forceinline__ void store_scaled4(_Float16* p, float4 v, float sc) {
    v4h h;
    h.x = (_Float16)(v.x * sc);
    h.y = (_Float16)(v.y * sc);
    h.z = (_Float16)(v.z * sc);
    h.w = (_Float16)(v.w * sc);
    *(v4h*)p = h;
}

__device__ __forceinline__ float sumsq4(float4 v) {
    return v.x * v.x + v.y * v.y + v.z * v.z + v.w * v.w;
}

// A-fragment (16x32 f16): lanes 0-15 hold K=k..k+15 of row m, lanes 16-31 hold K=k+16..k+31.
__device__ __forceinline__ v16h load_fragA(const _Float16* base, int r0, int kb,
                                           int m16, int ksel) {
    const _Float16* p = base + (r0 + m16) * LDS_STRIDE + kb + (ksel << 4);
    union { v16h v; v8h h[2]; } u;
    u.h[0] = *(const v8h*)p;
    u.h[1] = *(const v8h*)(p + 8);
    return u.v;
}

// B-fragment (32x16 f16): lanes 0-15 hold K=k..k+15 of col n, lanes 16-31 hold K=k+16..k+31.
__device__ __forceinline__ v16h load_fragB(const _Float16* base, int c0, int kb,
                                           int m16, int ksel) {
    const _Float16* p = base + (c0 + m16) * LDS_STRIDE + kb + (ksel << 4);
    union { v16h v; v8h h[2]; } u;
    u.h[0] = *(const v8h*)p;
    u.h[1] = *(const v8h*)(p + 8);
    return u.v;
}

__device__ __forceinline__ v8f wmma_f16(v16h a, v16h b, v8f c) {
    return __builtin_amdgcn_wmma_f32_16x16x32_f16(false, a, false, b,
                                                  (short)0, c, false, false);
}

extern "C" __global__ __launch_bounds__(256)
void subcenter_arcface_wmma(const float* __restrict__ x,
                            const float* __restrict__ W,
                            float* __restrict__ out) {
    extern __shared__ _Float16 smem[];
    _Float16* Alds = smem;
    _Float16* Blds = smem + A_HALVES;

    const int tid  = threadIdx.x;
    const int lane = tid & 31;
    const int w    = tid >> 5;

    // Grid: x = M tiles (4), y = class tiles (3125).
    // HSA dispatch walks x fastest, so the 4 workgroups sharing one W tile are
    // adjacent in dispatch order and share the 196KB W tile through the 192MB L2
    // -> W streams from HBM exactly once (~614MB total).
    const int row_base  = blockIdx.x * TILE_M;
    const int ncol_base = blockIdx.y * TILE_N;
    const int cls_base  = blockIdx.y * (TILE_N / KSUB);

    // ---- x tile: 16 rows per wave; normalize in-flight, convert to f16 ----
    for (int i = 0; i < 16; ++i) {
        const int r = (w << 4) + i;
        const float4* src = (const float4*)(x + (size_t)(row_base + r) * DIM);
        float4 v0 = src[lane];
        float4 v1 = src[lane + 32];
        float4 v2 = src[lane + 64];
        float4 v3 = src[lane + 96];
        float ss = sumsq4(v0) + sumsq4(v1) + sumsq4(v2) + sumsq4(v3);
        for (int off = 16; off; off >>= 1) ss += __shfl_xor(ss, off, 32);
        const float sc = __frsqrt_rn(ss);
        _Float16* dst = Alds + r * LDS_STRIDE;
        store_scaled4(dst + 4 * lane,       v0, sc);
        store_scaled4(dst + 128 + 4 * lane, v1, sc);
        store_scaled4(dst + 256 + 4 * lane, v2, sc);
        store_scaled4(dst + 384 + 4 * lane, v3, sc);
    }

    // ---- W tile: 12 subcenter rows per wave; row-norm via wave reduce, single read of W ----
    for (int i = 0; i < 12; ++i) {
        const int r = w * 12 + i;
        const float4* src = (const float4*)(W + (size_t)(ncol_base + r) * DIM);
        float4 v0 = src[lane];
        float4 v1 = src[lane + 32];
        float4 v2 = src[lane + 64];
        float4 v3 = src[lane + 96];
        float ss = sumsq4(v0) + sumsq4(v1) + sumsq4(v2) + sumsq4(v3);
        for (int off = 16; off; off >>= 1) ss += __shfl_xor(ss, off, 32);
        const float sc = __frsqrt_rn(ss);
        _Float16* dst = Blds + r * LDS_STRIDE;
        store_scaled4(dst + 4 * lane,       v0, sc);
        store_scaled4(dst + 128 + 4 * lane, v1, sc);
        store_scaled4(dst + 256 + 4 * lane, v2, sc);
        store_scaled4(dst + 384 + 4 * lane, v3, sc);
    }

    __syncthreads();

    // ---- WMMA main loop: each wave owns a 32x48 sub-tile (2x3 fragments) ----
    const int rfb  = (w >> 1) * 32;   // wave row-fragment base
    const int cfb  = (w & 1) * 48;    // wave col-fragment base
    const int m16  = lane & 15;
    const int ksel = lane >> 4;

    v8f acc[2][3] = {};
#pragma unroll
    for (int kk = 0; kk < 16; ++kk) {
        const int kb = kk * 32;
        v16h a0 = load_fragA(Alds, rfb,      kb, m16, ksel);
        v16h a1 = load_fragA(Alds, rfb + 16, kb, m16, ksel);
        v16h b0 = load_fragB(Blds, cfb,      kb, m16, ksel);
        v16h b1 = load_fragB(Blds, cfb + 16, kb, m16, ksel);
        v16h b2 = load_fragB(Blds, cfb + 32, kb, m16, ksel);
        acc[0][0] = wmma_f16(a0, b0, acc[0][0]);
        acc[0][1] = wmma_f16(a0, b1, acc[0][1]);
        acc[0][2] = wmma_f16(a0, b2, acc[0][2]);
        acc[1][0] = wmma_f16(a1, b0, acc[1][0]);
        acc[1][1] = wmma_f16(a1, b1, acc[1][1]);
        acc[1][2] = wmma_f16(a1, b2, acc[1][2]);
    }

    __syncthreads();  // done reading A/B from LDS; reuse the A region for C staging

    float* Clds = (float*)smem;
#pragma unroll
    for (int i = 0; i < 2; ++i)
#pragma unroll
        for (int j = 0; j < 3; ++j)
#pragma unroll
            for (int v = 0; v < 8; ++v) {
                const int m = rfb + i * 16 + (ksel << 3) + v;
                const int n = cfb + j * 16 + m16;
                Clds[m * C_STRIDE + n] = acc[i][j][v];
            }

    __syncthreads();

    // ---- epilogue: max over 3 subcenters + S*cos(arccos(c)+M), coalesced stores ----
    const float cM = 0.8775825618903728f;   // cos(0.5)
    const float sM = 0.4794255386042030f;   // sin(0.5)
#pragma unroll
    for (int t = 0; t < 16; ++t) {
        const int lin = t * 256 + tid;      // 128 rows x 32 classes = 4096
        const int m = lin >> 5;
        const int o = lin & 31;
        const float* c3 = Clds + m * C_STRIDE + 3 * o;
        float c = fmaxf(c3[0], fmaxf(c3[1], c3[2]));
        c = fminf(1.0f, fmaxf(-1.0f, c));
        const float s = sqrtf(fmaxf(0.0f, 1.0f - c * c));
        out[(size_t)(row_base + m) * OUT_CLASSES + cls_base + o] =
            64.0f * (c * cM - s * sM);
    }
}

extern "C" void kernel_launch(void* const* d_in, const int* in_sizes, int n_in,
                              void* d_out, int out_size, void* d_ws, size_t ws_size,
                              hipStream_t stream) {
    const float* x = (const float*)d_in[0];   // [512, 512] f32
    const float* W = (const float*)d_in[1];   // [100000, 3, 512] f32
    float* out = (float*)d_out;               // [512, 100000] f32

    dim3 grid(512 / TILE_M, NCOL / TILE_N, 1);   // (4, 3125): x-siblings share W tile in L2
    const size_t shmem = (size_t)(A_HALVES + B_HALVES) * sizeof(_Float16); // 232,960 B
    hipLaunchKernelGGL(subcenter_arcface_wmma, grid, dim3(256, 1, 1), shmem, stream,
                       x, W, out);
}
